// LocallyConnected1d_57174604644926
// MI455X (gfx1250) — compile-verified, matrix operand order checked by reading, not run on hardware
//
#include <hip/hip_runtime.h>

typedef __attribute__((ext_vector_type(16))) _Float16 v16h;
typedef __attribute__((ext_vector_type(8)))  float    v8f;

// Problem constants (from reference): x (64,64,512) f32, w (1,64,64,504,9) f32,
// out (64,64,504) f32.  out[b,o,l] = sum_{i,k} x[b,i,l+k] * w[o,i,l,k]
constexpr int Bk   = 64;            // batch
constexpr int Ok   = 64;            // output channels
constexpr int Ik   = 64;            // input channels
constexpr int KW   = 9;             // kernel width
constexpr int LIN  = 512;
constexpr int LOUT = 504;
constexpr int KD   = Ik * KW;       // 576 = reduction dimension
constexpr int KPAD = KD + 16;       // 592 halfs -> 1184B row stride: 32B-aligned, de-phases banks
constexpr int NTHREADS = 512;       // 16 wave32 per workgroup
constexpr int KCH  = 32;            // WMMA K per instruction (f16)
constexpr int NCH  = KD / KCH;      // 18 WMMA steps per tile

__global__ __launch_bounds__(NTHREADS)
void lc1d_wmma_kernel(const float* __restrict__ x,
                      const float* __restrict__ w,
                      float* __restrict__ out)
{
  extern __shared__ __align__(128) char smem[];
  _Float16* __restrict__ xa = (_Float16*)smem;       // [Bk][KPAD]  A: x-window, f16
  _Float16* __restrict__ wb = xa + Bk * KPAD;        // [Ok][KPAD]  B: weights,  f16

  const int l   = blockIdx.x;      // one workgroup per output position l
  const int tid = threadIdx.x;

  // ---- Stage A matrix: xa[b][i*9+k] = (f16) x[b][i][l+k] ----
  // Per (b,i): 9 contiguous f32 in global -> 9 contiguous f16 in LDS.
  for (int p = tid; p < Bk * Ik; p += NTHREADS) {
    const int b = p >> 6, i = p & 63;
    const float* src = x + ((size_t)b * Ik + i) * LIN + l;
    _Float16* dst = xa + b * KPAD + i * KW;
#pragma unroll
    for (int k = 0; k < KW; ++k) dst[k] = (_Float16)src[k];
  }
  // ---- Stage B matrix: wb[o][i*9+k] = (f16) w[o][i][l][k] ----
  for (int p = tid; p < Ok * Ik; p += NTHREADS) {
    const int o = p >> 6, i = p & 63;
    const float* src = w + (((size_t)o * Ik + i) * LOUT + l) * KW;
    _Float16* dst = wb + o * KPAD + i * KW;
#pragma unroll
    for (int k = 0; k < KW; ++k) dst[k] = (_Float16)src[k];
  }
  __syncthreads();

  // ---- Compute: each wave owns one 16x16 (b-tile, o-tile) of the 64x64 output ----
  const int wave = tid >> 5;          // 0..15
  const int lane = tid & 31;
  const int bt   = wave >> 2;         // 4 b-tiles
  const int ot   = wave & 3;          // 4 o-tiles
  const int lgrp = lane >> 4;         // 0: K=0..15, 1: K=16..31 (contiguous K order used here for both A and B)
  const int lrow = lane & 15;         // A: row M, B: column N

  // A fragment lane view: row M = bt*16+lrow, 16 consecutive K starting at lgrp*16
  const _Float16* arow = xa + (bt * 16 + lrow) * KPAD + lgrp * 16;
  // B fragment lane view: col N = ot*16+lrow, same K slice (wb stored [N][K])
  const _Float16* brow = wb + (ot * 16 + lrow) * KPAD + lgrp * 16;

  v8f acc = {};
#pragma unroll 3
  for (int c = 0; c < NCH; ++c) {
    v16h a = *(const v16h*)(arow + c * KCH);   // 32B-aligned: 2x ds_load_b128
    v16h b = *(const v16h*)(brow + c * KCH);
    acc = __builtin_amdgcn_wmma_f32_16x16x32_f16(
        /*neg_a=*/false, a, /*neg_b=*/false, b,
        /*c_mod=*/(short)0, acc, /*reuse_a=*/false, /*reuse_b=*/false);
  }

  // ---- Store C: lane holds N = lrow, rows M = r + 8*lgrp (documented f32 C/D layout) ----
  float* obase = out + (((size_t)bt * 16) * Ok + (ot * 16 + lrow)) * LOUT + l;
#pragma unroll
  for (int r = 0; r < 8; ++r) {
    const int m = r + 8 * lgrp;
    obase[(size_t)m * Ok * LOUT] = acc[r];
  }
}

extern "C" void kernel_launch(void* const* d_in, const int* in_sizes, int n_in,
                              void* d_out, int out_size, void* d_ws, size_t ws_size,
                              hipStream_t stream) {
  const float* x = (const float*)d_in[0];   // 64*64*512 f32
  const float* w = (const float*)d_in[1];   // 1*64*64*504*9 f32
  float* out = (float*)d_out;               // 64*64*504 f32

  const size_t shmem = (size_t)(Bk + Ok) * KPAD * sizeof(_Float16); // ~148 KB, fits 320 KB WGP LDS
  dim3 grid(LOUT);        // 504 workgroups, one per l
  dim3 block(NTHREADS);   // 16 wave32
  lc1d_wmma_kernel<<<grid, block, shmem, stream>>>(x, w, out);
}
